// attention_prediction_asym_14439680049267
// MI455X (gfx1250) — compile-verified, matrix operand order checked by reading, not run on hardware
//
#include <hip/hip_runtime.h>
#include <hip/hip_bf16.h>

typedef __attribute__((ext_vector_type(16))) _Float16 v16h;
typedef __attribute__((ext_vector_type(8)))  float    v8f;

#define FEAT    128
#define L1OUT   16
#define L2D     17
#define WPB     8      // waves per block (256 threads, wave32)
#define BATCHES 4      // 32 edges per batch per wave -> 1024 edges per block

// ---- float <-> order-preserving uint (for atomicMax on floats) ----
__device__ __forceinline__ unsigned ordf(float f) {
  unsigned u = __float_as_uint(f);
  return (u & 0x80000000u) ? ~u : (u | 0x80000000u);
}
__device__ __forceinline__ float unordf(unsigned o) {
  unsigned u = (o & 0x80000000u) ? (o ^ 0x80000000u) : ~o;
  return __uint_as_float(u);
}
#define ORD_NEG_INF 0x007FFFFFu   // ordf(-inf)

__device__ __forceinline__ float leaky(float x) { return x >= 0.0f ? x : 0.1f * x; }

// ---------------- K0: init per-node accumulators ----------------
__global__ void k_init(float* deg, unsigned* mord, float* ssum, int N) {
  int n = blockIdx.x * blockDim.x + threadIdx.x;
  if (n < N) { deg[n] = 0.0f; mord[n] = ORD_NEG_INF; ssum[n] = 0.0f; }
}

// ---------------- K1: deg_abs[i] = segment_sum(|v|) ----------------
__global__ void k_deg(const int* __restrict__ ind, const float* __restrict__ val,
                      float* deg, int E) {
  int e = blockIdx.x * blockDim.x + threadIdx.x;
  if (e < E) atomicAdd(&deg[ind[e]], fabsf(val[e]));
}

// ---------------- K2: WMMA edge MLP + segment max ----------------
__global__ __launch_bounds__(256) void
k_mlp(const int* __restrict__ ind, const float* __restrict__ val,
      const float* __restrict__ f,
      const float* __restrict__ W1, const float* __restrict__ b1,
      const float* __restrict__ W2, const float* __restrict__ b2,
      const float* __restrict__ Wc, const float* __restrict__ bc,
      const float* __restrict__ deg, unsigned* mord, float* evals, int E)
{
  __shared__ float sW2[L2D * L2D];
  __shared__ float sb2[L2D];
  __shared__ float sWc[L2D];
  __shared__ float sbc;
  __shared__ float sY[WPB][32][L2D + 1];   // 32 edges x (16 feats + norm), padded

  const int tid = threadIdx.x;
  for (int t = tid; t < L2D * L2D; t += 256) sW2[t] = W2[t];
  if (tid < L2D) { sb2[tid] = b2[tid]; sWc[tid] = Wc[tid]; }
  if (tid == 0) sbc = bc[0];
  __syncthreads();

  const int wave = tid >> 5;
  const int lane = tid & 31;
  const int r    = lane & 15;   // row (A) / col N (B, C)
  const int half = lane >> 4;   // lane half selects K sub-block

  // --- B fragments: W1 (256x16, row-major) as 8 K-chunks of f16, kept in VGPRs ---
  // 16-bit operand layout: element p -> K = (p/8)*16 + half*8 + (p%8), N = r
  v16h Bfrag[8];
  #pragma unroll
  for (int c = 0; c < 8; ++c) {
    #pragma unroll
    for (int p = 0; p < 16; ++p) {
      int k = 32 * c + ((p >> 3) << 4) + (half << 3) + (p & 7);
      Bfrag[c][p] = (_Float16)W1[k * L1OUT + r];
    }
  }
  const float b1r = b1[r];

  const int waveBase = (blockIdx.x * WPB + wave) * (32 * BATCHES);

  for (int bt = 0; bt < BATCHES; ++bt) {
    const int base = waveBase + 32 * bt;

    // edge handled by this lane in phase 2 (and for the norm feature)
    const int e  = base + lane;
    const int ec = e < E ? e : E - 1;
    const int ie = ind[ec];
    const float nv = fabsf(val[ec]) / deg[ie];
    sY[wave][lane][16] = nv;

    // --- two 16-edge tiles: y = leaky(concat(f[i],f[j]) @ W1 + b1) via WMMA ---
    #pragma unroll
    for (int t = 0; t < 2; ++t) {
      const int eA  = base + 16 * t + r;
      const int eAc = eA < E ? eA : E - 1;
      const int ii = ind[eAc];
      const int jj = ind[E + eAc];
      const float* __restrict__ fi = f + (size_t)ii * FEAT;
      const float* __restrict__ fj = f + (size_t)jj * FEAT;

      v8f acc = {0.f, 0.f, 0.f, 0.f, 0.f, 0.f, 0.f, 0.f};
      #pragma unroll
      for (int c = 0; c < 8; ++c) {
        const float* src = (c < 4) ? fi : fj;
        const int b0 = ((c & 3) << 5) + (half << 3);     // 32-byte aligned
        const float4 u0 = *(const float4*)(src + b0);
        const float4 u1 = *(const float4*)(src + b0 + 4);
        const float4 u2 = *(const float4*)(src + b0 + 16);
        const float4 u3 = *(const float4*)(src + b0 + 20);
        v16h a;
        a[0]=(_Float16)u0.x;  a[1]=(_Float16)u0.y;  a[2]=(_Float16)u0.z;  a[3]=(_Float16)u0.w;
        a[4]=(_Float16)u1.x;  a[5]=(_Float16)u1.y;  a[6]=(_Float16)u1.z;  a[7]=(_Float16)u1.w;
        a[8]=(_Float16)u2.x;  a[9]=(_Float16)u2.y;  a[10]=(_Float16)u2.z; a[11]=(_Float16)u2.w;
        a[12]=(_Float16)u3.x; a[13]=(_Float16)u3.y; a[14]=(_Float16)u3.z; a[15]=(_Float16)u3.w;
        acc = __builtin_amdgcn_wmma_f32_16x16x32_f16(
            /*neg_a=*/false, a, /*neg_b=*/false, Bfrag[c],
            /*c_mod=*/(short)0, acc, /*reuse_a=*/false, /*reuse_b=*/false);
      }
      // C layout: lane holds column n=r, edges M = v + 8*half
      #pragma unroll
      for (int v = 0; v < 8; ++v) {
        float y = leaky(acc[v] + b1r);
        sY[wave][16 * t + 8 * half + v][r] = y;
      }
    }

    // per-wave LDS: ensure tile stores complete before cross-lane reads
    asm volatile("s_wait_dscnt 0x0" ::: "memory");

    // --- phase 2: per-lane 17-dim layer 2 + collapse for edge e ---
    float x2[L2D];
    #pragma unroll
    for (int k = 0; k < L2D; ++k) x2[k] = sY[wave][lane][k];
    float z[L2D];
    #pragma unroll
    for (int n = 0; n < L2D; ++n) z[n] = sb2[n];
    #pragma unroll
    for (int k = 0; k < L2D; ++k) {
      const float xk = x2[k];
      #pragma unroll
      for (int n = 0; n < L2D; ++n) z[n] = fmaf(xk, sW2[k * L2D + n], z[n]);
    }
    float ev = sbc;
    #pragma unroll
    for (int n = 0; n < L2D; ++n) ev = fmaf(leaky(z[n]), sWc[n], ev);

    if (e < E) {
      evals[e] = ev;
      atomicMax(&mord[ie], ordf(ev));
    }
    asm volatile("s_wait_dscnt 0x0" ::: "memory");
  }
}

// ---------------- K3: ex = exp(e - m[i]); s[i] += ex ----------------
__global__ void k_exp(const int* __restrict__ ind, const float* __restrict__ evals,
                      const unsigned* __restrict__ mord, float* ssum,
                      float* out, int E) {
  int e = blockIdx.x * blockDim.x + threadIdx.x;
  if (e < E) {
    int i = ind[e];
    float ex = expf(evals[e] - unordf(mord[i]));
    out[e] = ex;
    atomicAdd(&ssum[i], ex);
  }
}

// ---------------- K4: out[e] /= s[i] ----------------
__global__ void k_norm(const int* __restrict__ ind, const float* __restrict__ ssum,
                       float* out, int E) {
  int e = blockIdx.x * blockDim.x + threadIdx.x;
  if (e < E) out[e] = out[e] / ssum[ind[e]];
}

extern "C" void kernel_launch(void* const* d_in, const int* in_sizes, int n_in,
                              void* d_out, int out_size, void* d_ws, size_t ws_size,
                              hipStream_t stream) {
  const int*   ind = (const int*)d_in[0];    // Jt_ind  [2,E]
  const float* val = (const float*)d_in[1];  // Jt_val  [E]
  const float* f   = (const float*)d_in[2];  // f       [N,128]
  const float* W1  = (const float*)d_in[4];  // [256,16]
  const float* b1  = (const float*)d_in[5];  // [16]
  const float* W2  = (const float*)d_in[6];  // [17,17]
  const float* b2  = (const float*)d_in[7];  // [17]
  const float* Wc  = (const float*)d_in[8];  // [17,1]
  const float* bc  = (const float*)d_in[9];  // [1]
  float* out = (float*)d_out;

  const int E = in_sizes[1];
  const int N = in_sizes[2] / FEAT;

  // workspace layout: deg[N] | mord[N] | ssum[N] | evals[E]
  float*    deg   = (float*)d_ws;
  unsigned* mord  = (unsigned*)(deg + N);
  float*    ssum  = (float*)(mord + N);
  float*    evals = ssum + N;

  k_init<<<(N + 255) / 256, 256, 0, stream>>>(deg, mord, ssum, N);
  k_deg <<<(E + 255) / 256, 256, 0, stream>>>(ind, val, deg, E);
  k_mlp <<<(E + (WPB * BATCHES * 32) - 1) / (WPB * BATCHES * 32), 256, 0, stream>>>(
      ind, val, f, W1, b1, W2, b2, Wc, bc, deg, mord, evals, E);
  k_exp <<<(E + 255) / 256, 256, 0, stream>>>(ind, evals, mord, ssum, out, E);
  k_norm<<<(E + 255) / 256, 256, 0, stream>>>(ind, ssum, out, E);
}
